// FastenRGAT_22196390986275
// MI455X (gfx1250) — matrix operand order repaired and checked by reading, not compiled, not run on hardware
//
#include <hip/hip_runtime.h>
#include <hip/hip_bf16.h>
#include <math.h>

// ---------------------------------------------------------------------------
// RGAT (2 conv layers + linear head + log_softmax) for MI455X / gfx1250.
// GEMMs (x @ W[r]) use v_wmma_f32_16x16x32_f16; edge softmax/scatter stays
// f32 (bandwidth/L2-atomic bound; h table of 102MB is L2-resident on 192MB L2).
// ---------------------------------------------------------------------------

typedef __attribute__((ext_vector_type(16))) _Float16 v16h;
typedef __attribute__((ext_vector_type(8)))  float    v8f;

union AFragU { v16h v; _Float16 h[16]; };
union DFragU { v8f  v; float     f[8];  };

#define LEAKY_SLOPE 0.2f

// --------------------------- per-relation transform -------------------------
// h[r,n,c] = sum_d x[n,d] * W[r,d,c]   (WMMA 16x16x32 f16, f32 accumulate)
// grid: (ceil(n/128), R), block: 256 (8 waves, each wave owns a 16-row tile)
__global__ __launch_bounds__(256)
void rgat_transform_kernel(const float* __restrict__ X,
                           const float* __restrict__ W,
                           float* __restrict__ H, int n) {
  const int r = blockIdx.y;
  // W[r] transposed into LDS as f16: Wl[c][k], padded row to 72 halfs.
  __shared__ _Float16 Wl[64][72];
  const float* Wr = W + (size_t)r * 64 * 64;
  for (int idx = threadIdx.x; idx < 64 * 64; idx += blockDim.x) {
    const int k = idx >> 6, c = idx & 63;
    Wl[c][k] = (_Float16)Wr[idx];           // Wr[k*64 + c]
  }
  __syncthreads();

  const int wave = threadIdx.x >> 5;
  const int lane = threadIdx.x & 31;
  const int hh   = lane >> 4;               // half-wave select
  const int lm   = lane & 15;
  const int rowBase = (blockIdx.x * 8 + wave) * 16;
  if (rowBase >= n) return;                 // wave-uniform: EXEC stays all-1s
  const bool fullTile = (rowBase + 16 <= n);

  // A fragments (ISA 16-bit A 16x32 layout): lanes 0-15 rows, K {0-7,16-23};
  // lanes 16-31 same rows, K {8-15,24-31}. a1 covers K+32.
  AFragU a0, a1;
  const int row = rowBase + lm;
  if (fullTile || row < n) {
    const float* xr = X + (size_t)row * 64 + (hh ? 8 : 0);
#pragma unroll
    for (int j = 0; j < 8; ++j) {           // 8 contiguous floats per group
      a0.h[j]     = (_Float16)xr[j];
      a0.h[8 + j] = (_Float16)xr[16 + j];
      a1.h[j]     = (_Float16)xr[32 + j];
      a1.h[8 + j] = (_Float16)xr[48 + j];
    }
  } else {
#pragma unroll
    for (int j = 0; j < 16; ++j) { a0.h[j] = (_Float16)0.f; a1.h[j] = (_Float16)0.f; }
  }

#pragma unroll
  for (int ct = 0; ct < 4; ++ct) {          // four 16-col output tiles
    const int col = ct * 16 + lm;
    // B fragments: element j -> K = kc*32 + (hh?8:0) + j + (j>=8?8:0), column = col.
    AFragU b0, b1;
#pragma unroll
    for (int j = 0; j < 16; ++j) {
      b0.h[j] = Wl[col][(hh ? 8 : 0) + j + (j >= 8 ? 8 : 0)];
      b1.h[j] = Wl[col][32 + (hh ? 8 : 0) + j + (j >= 8 ? 8 : 0)];
    }
    v8f acc = {};
    acc = __builtin_amdgcn_wmma_f32_16x16x32_f16(false, a0.v, false, b0.v,
                                                 (short)0, acc, false, false);
    acc = __builtin_amdgcn_wmma_f32_16x16x32_f16(false, a1.v, false, b1.v,
                                                 (short)0, acc, false, false);
    DFragU d; d.v = acc;
    // D layout: VGPR v -> M = v + 8*hh, lane -> N = lm. One base pointer,
    // constant 64-float stride: unguarded coalesced stores on the fast path.
    float* outp = H + ((size_t)r * n + rowBase + (hh ? 8 : 0)) * 64 + ct * 16 + lm;
    if (fullTile) {
#pragma unroll
      for (int v = 0; v < 8; ++v) outp[(size_t)v * 64] = d.f[v];
    } else {
#pragma unroll
      for (int v = 0; v < 8; ++v)
        if (rowBase + (hh ? 8 : 0) + v < n) outp[(size_t)v * 64] = d.f[v];
    }
  }
}

// ------------------------ qn / kn attention logits ---------------------------
// one wave per (r, node): 64-feature dot, shfl-reduced (wave32).
__global__ __launch_bounds__(256)
void rgat_qk_kernel(const float* __restrict__ H, const float* __restrict__ q,
                    const float* __restrict__ k, float* __restrict__ QN,
                    float* __restrict__ KN, int n, int R) {
  const int wid  = (int)((blockIdx.x * blockDim.x + threadIdx.x) >> 5);
  const int lane = threadIdx.x & 31;
  if (wid >= R * n) return;
  const int r = wid / n;
  const float* h = H + (size_t)wid * 64;
  const float h0 = h[lane], h1 = h[lane + 32];
  float qs = h0 * q[r * 64 + lane] + h1 * q[r * 64 + lane + 32];
  float ks = h0 * k[r * 64 + lane] + h1 * k[r * 64 + lane + 32];
#pragma unroll
  for (int m = 16; m; m >>= 1) {
    qs += __shfl_xor(qs, m, 32);
    ks += __shfl_xor(ks, m, 32);
  }
  if (lane == 0) { QN[wid] = qs; KN[wid] = ks; }
}

// ------------------------------ init scratch --------------------------------
__global__ void rgat_init_kernel(float* amax, float* denom, float* acc, int n) {
  const int i = blockIdx.x * blockDim.x + threadIdx.x;
  if (i < n) { amax[i] = -__builtin_inff(); denom[i] = 0.f; }
  if (i < n * 64) acc[i] = 0.f;
}

// ------------- pass 1: edge logit + leaky relu + segment max ----------------
__global__ void rgat_edge_max_kernel(const int* __restrict__ src,
                                     const int* __restrict__ dst,
                                     const int* __restrict__ et,
                                     const float* __restrict__ QN,
                                     const float* __restrict__ KN,
                                     float* __restrict__ A, float* amax,
                                     int E, int n) {
  const int e = blockIdx.x * blockDim.x + threadIdx.x;
  if (e >= E) return;
  const int t = et[e], s = src[e], d = dst[e];
  float a = QN[t * n + d] + KN[t * n + s];
  a = a > 0.f ? a : LEAKY_SLOPE * a;
  A[e] = a;
  // monotonic float-max via signed/unsigned integer atomics (-inf init)
  if (a >= 0.f) atomicMax((int*)(amax + d), __float_as_int(a));
  else          atomicMin((unsigned int*)(amax + d), __float_as_uint(a));
}

// -------------------- pass 2: exp + segment sum (denominator) ---------------
__global__ void rgat_edge_exp_kernel(const int* __restrict__ dst,
                                     const float* __restrict__ A,
                                     const float* __restrict__ amax,
                                     float* __restrict__ EXb, float* denom,
                                     int E) {
  const int e = blockIdx.x * blockDim.x + threadIdx.x;
  if (e >= E) return;
  const int d = dst[e];
  float m = amax[d];
  if (m == -__builtin_inff()) m = 0.f;      // reference's isfinite fix-up
  const float ex = __expf(A[e] - m);
  EXb[e] = ex;
  atomicAdd(denom + d, ex);
}

// --------- pass 3: message scatter  ACC[dst] += alpha * h[et,src] -----------
// one wave per edge slot, grid-stride; lane covers features c and c+32.
// Next edge's h-row is prefetched (global_prefetch_b8) while this edge's
// L2-resident f32 atomics are in flight.
__global__ __launch_bounds__(256)
void rgat_edge_scatter_kernel(const int* __restrict__ src,
                              const int* __restrict__ dst,
                              const int* __restrict__ et,
                              const float* __restrict__ H,
                              const float* __restrict__ EXb,
                              const float* __restrict__ denom,
                              float* __restrict__ ACC, int E, int n) {
  const int lane = threadIdx.x & 31;
  const int nw   = (int)((gridDim.x * blockDim.x) >> 5);
  int e = (int)((blockIdx.x * blockDim.x + threadIdx.x) >> 5);
  if (e >= E) return;
  int t = et[e], s = src[e], d = dst[e];
  while (true) {
    const float* hs = H + ((size_t)t * n + s) * 64;
    const float alpha = EXb[e] / (denom[d] + 1e-16f);
    const int en = e + nw;
    int tn = 0, sn = 0, dn = 0;
    if (en < E) {                           // carry next indices + prefetch row
      tn = et[en]; sn = src[en]; dn = dst[en];
      __builtin_prefetch(H + ((size_t)tn * n + sn) * 64 + lane, 0, 0);
    }
    atomicAdd(ACC + (size_t)d * 64 + lane,      alpha * hs[lane]);
    atomicAdd(ACC + (size_t)d * 64 + lane + 32, alpha * hs[lane + 32]);
    if (en >= E) break;
    e = en; t = tn; s = sn; d = dn;
  }
}

// ----------------------------- bias + relu ----------------------------------
__global__ void rgat_bias_relu_kernel(const float* __restrict__ ACC,
                                      const float* __restrict__ b,
                                      float* __restrict__ out, int n) {
  const int i = blockIdx.x * blockDim.x + threadIdx.x;
  if (i >= n * 64) return;
  const float v = ACC[i] + b[i & 63];
  out[i] = v > 0.f ? v : 0.f;
}

// ---------------- head: logits = h @ lin_w + lin_b ; log_softmax ------------
// one wave per node; lane owns outputs f=lane and f=lane+32 (lanes 0-7 only).
__global__ __launch_bounds__(256)
void rgat_head_kernel(const float* __restrict__ Hin,
                      const float* __restrict__ Wl,
                      const float* __restrict__ bl,
                      float* __restrict__ out, int n, int OUTF) {
  const int i    = (int)((blockIdx.x * blockDim.x + threadIdx.x) >> 5);
  const int lane = threadIdx.x & 31;
  if (i >= n) return;
  const float* h = Hin + (size_t)i * 64;
  const int f1 = lane, f2 = lane + 32;
  const bool has1 = f1 < OUTF;
  const bool has2 = f2 < OUTF;
  const int wf1 = has1 ? f1 : 0;            // clamped indices: branchless loop
  const int wf2 = has2 ? f2 : 0;
  float l1 = has1 ? bl[wf1] : 0.f;
  float l2 = has2 ? bl[wf2] : 0.f;
  for (int kk = 0; kk < 64; ++kk) {
    const float hv = h[kk];
    l1 += hv * Wl[kk * OUTF + wf1];
    l2 += hv * Wl[kk * OUTF + wf2];
  }
  float mm = has1 ? l1 : -__builtin_inff();
  if (has2) mm = fmaxf(mm, l2);
#pragma unroll
  for (int m = 16; m; m >>= 1) mm = fmaxf(mm, __shfl_xor(mm, m, 32));
  float ssum = (has1 ? __expf(l1 - mm) : 0.f) + (has2 ? __expf(l2 - mm) : 0.f);
#pragma unroll
  for (int m = 16; m; m >>= 1) ssum += __shfl_xor(ssum, m, 32);
  const float lg = __logf(ssum);
  if (has1) out[(size_t)i * OUTF + f1] = l1 - mm - lg;
  if (has2) out[(size_t)i * OUTF + f2] = l2 - mm - lg;
}

// ---------------------------------------------------------------------------
extern "C" void kernel_launch(void* const* d_in, const int* in_sizes, int n_in,
                              void* d_out, int out_size, void* d_ws, size_t ws_size,
                              hipStream_t stream) {
  (void)n_in; (void)ws_size; (void)out_size;
  const float* x  = (const float*)d_in[0];
  const int*   ei = (const int*)d_in[1];
  const int*   et = (const int*)d_in[2];
  // d_in[3] = tensor_slice (fasten metadata, math-irrelevant)
  const float* W1 = (const float*)d_in[4];
  const float* q1 = (const float*)d_in[5];
  const float* k1 = (const float*)d_in[6];
  const float* b1 = (const float*)d_in[7];
  const float* W2 = (const float*)d_in[8];
  const float* q2 = (const float*)d_in[9];
  const float* k2 = (const float*)d_in[10];
  const float* b2 = (const float*)d_in[11];
  const float* lw = (const float*)d_in[12];
  const float* lb = (const float*)d_in[13];

  const int n = in_sizes[0] / 64;
  const int E = in_sizes[2];
  const int R = in_sizes[5] / 64;
  const int OUTF = in_sizes[13];
  const int* src = ei;
  const int* dst = ei + E;

  // workspace layout (floats)
  float* ws = (float*)d_ws;
  size_t off = 0;
  float* H     = ws + off; off += (size_t)R * n * 64;   // per-relation transforms
  float* QN    = ws + off; off += (size_t)R * n;
  float* KN    = ws + off; off += (size_t)R * n;
  float* Abuf  = ws + off; off += (size_t)E;            // edge logits
  float* EXb   = ws + off; off += (size_t)E;            // edge exp
  float* amax  = ws + off; off += (size_t)n;
  float* denom = ws + off; off += (size_t)n;
  float* ACC   = ws + off; off += (size_t)n * 64;       // message accumulator
  float* HMID  = ws + off; off += (size_t)n * 64;       // layer activations

  const int scatterBlocks = 8192;                       // grid-stride waves

  auto run_layer = [&](const float* xin, const float* Wm, const float* qv,
                       const float* kv, const float* bv, float* xout) {
    dim3 gT((unsigned)((n + 127) / 128), (unsigned)R);
    rgat_transform_kernel<<<gT, 256, 0, stream>>>(xin, Wm, H, n);
    rgat_qk_kernel<<<(R * n * 32 + 255) / 256, 256, 0, stream>>>(H, qv, kv, QN, KN, n, R);
    rgat_init_kernel<<<(n * 64 + 255) / 256, 256, 0, stream>>>(amax, denom, ACC, n);
    rgat_edge_max_kernel<<<(E + 255) / 256, 256, 0, stream>>>(src, dst, et, QN, KN, Abuf, amax, E, n);
    rgat_edge_exp_kernel<<<(E + 255) / 256, 256, 0, stream>>>(dst, Abuf, amax, EXb, denom, E);
    rgat_edge_scatter_kernel<<<scatterBlocks, 256, 0, stream>>>(src, dst, et, H, EXb, denom, ACC, E, n);
    rgat_bias_relu_kernel<<<(n * 64 + 255) / 256, 256, 0, stream>>>(ACC, bv, xout, n);
  };

  run_layer(x,    W1, q1, k1, b1, HMID);   // layer 1: reads x, writes HMID
  run_layer(HMID, W2, q2, k2, b2, HMID);   // layer 2: in-place is safe (stream order)

  rgat_head_kernel<<<(n * 32 + 255) / 256, 256, 0, stream>>>(HMID, lw, lb, (float*)d_out, n, OUTF);
}
